// QLSTMHybrid_65481071398034
// MI455X (gfx1250) — matrix operand order repaired and checked by reading, not compiled, not run on hardware
//
#include <hip/hip_runtime.h>
#include <hip/hip_bf16.h>
#include <math.h>

// Problem dims
#define S_   1024
#define B_   256
#define D_   128
#define H_   256
#define A_   256
#define KTOT 384        // D + H
#define NF   1024       // 4 * H fused gate columns
#define NCH  8          // attention s-chunks
#define SCH  (S_ / NCH) // 128
#define NBLK 64         // persistent LSTM blocks (must all be co-resident)

typedef __attribute__((ext_vector_type(16))) __bf16 v16bf;
typedef __attribute__((ext_vector_type(8)))  __bf16 v8bf;
typedef __attribute__((ext_vector_type(8)))  float  v8f;
typedef __attribute__((ext_vector_type(4)))  unsigned int u32x4;
typedef __attribute__((ext_vector_type(8)))  int i32x8;
typedef __attribute__((ext_vector_type(4)))  int i32x4;

// ---- WMMA helpers -----------------------------------------------------------

__device__ __forceinline__ v8f wmma_bf16(v16bf a, v16bf b, v8f c) {
  return __builtin_amdgcn_wmma_f32_16x16x32_bf16(
      /*neg_a=*/false, a, /*neg_b=*/false, b,
      /*c_mod=*/(short)0, c, /*reuse_a=*/false, /*reuse_b=*/false);
}

// A-fragment (16x32 bf16) from row-major bf16 matrix.
// lanes 0-15: M=lane, K in [0,8)∪[16,24); lanes 16-31: M=lane-16, K in [8,16)∪[24,32)
__device__ __forceinline__ v16bf load_a_bf16(const __bf16* base, int row_stride, int lane) {
  const int r    = lane & 15;
  const int koff = (lane & 16) ? 8 : 0;
  const __bf16* p = base + (size_t)r * row_stride + koff;
  v8bf lo = *(const v8bf*)(p);
  v8bf hi = *(const v8bf*)(p + 16);
  return __builtin_shufflevector(lo, hi, 0,1,2,3,4,5,6,7,8,9,10,11,12,13,14,15);
}

// Same A-fragment but f32 source (convert in registers).
__device__ __forceinline__ v16bf load_a_f32(const float* base, int row_stride, int lane) {
  const int r    = lane & 15;
  const int koff = (lane & 16) ? 8 : 0;
  const float* p = base + (size_t)r * row_stride + koff;
  v16bf a;
#pragma unroll
  for (int j = 0; j < 8; ++j) a[j]     = (__bf16)p[j];
#pragma unroll
  for (int j = 0; j < 8; ++j) a[j + 8] = (__bf16)p[16 + j];
  return a;
}

// B-fragment (32x16 bf16) from row-major W^T (rows = N, KTOT wide).
// N = lane&15; lanes 0-15 take K [0,8)∪[16,24), lanes 16-31 take K [8,16)∪[24,32).
__device__ __forceinline__ v16bf load_b_rowmajT(const __bf16* Wt, int ktot, int n0,
                                                int kb, int lane) {
  const __bf16* p = Wt + (size_t)(n0 + (lane & 15)) * ktot + kb * 32 + ((lane & 16) ? 8 : 0);
  v8bf lo = *(const v8bf*)(p);
  v8bf hi = *(const v8bf*)(p + 16);
  return __builtin_shufflevector(lo, hi, 0,1,2,3,4,5,6,7,8,9,10,11,12,13,14,15);
}

__device__ __forceinline__ float sigmoidf_(float x) {
  return 1.f / (1.f + __expf(-x));
}

// ---- conversion / packing kernels ------------------------------------------

__global__ void cvt_x_kernel(const float* __restrict__ x, __bf16* __restrict__ xb, long n) {
  long i = (long)blockIdx.x * blockDim.x + threadIdx.x;
  if (i < n) xb[i] = (__bf16)x[i];
}

// Fuse Wf|Wi|Wg|Wo into bf16 W^T: Wp[n][k], n = gate*H + h.
__global__ void pack_gates_kernel(const float* __restrict__ Wf, const float* __restrict__ Wi,
                                  const float* __restrict__ Wg, const float* __restrict__ Wo,
                                  __bf16* __restrict__ Wp) {
  int i = blockIdx.x * blockDim.x + threadIdx.x;
  if (i >= NF * KTOT) return;
  int n = i / KTOT, k = i % KTOT;
  int g = n >> 8, h = n & 255;
  const float* W = (g == 0) ? Wf : (g == 1) ? Wi : (g == 2) ? Wg : Wo;
  Wp[(size_t)n * KTOT + k] = (__bf16)W[(size_t)k * H_ + h];
}

__global__ void pack_wa_kernel(const float* __restrict__ Wa, __bf16* __restrict__ Wpa) {
  int i = blockIdx.x * blockDim.x + threadIdx.x;
  if (i >= A_ * H_) return;
  int a = i / H_, k = i % H_;
  Wpa[(size_t)a * H_ + k] = (__bf16)Wa[(size_t)k * A_ + a];
}

__global__ void init_state_kernel(__bf16* __restrict__ h0, unsigned* __restrict__ bar) {
  int i = blockIdx.x * blockDim.x + threadIdx.x;
  if (i < B_ * H_) h0[i] = (__bf16)0.f;
  if (i < 2) bar[i] = 0u;
}

// ---- persistent LSTM: all 1024 steps in one launch --------------------------
// Grid: 64 blocks (bg = blockIdx>>4 picks 64 b-rows, ht = blockIdx&15 picks 16 h-cols).
// Block: 8 waves; wave w owns b-tile (w&3) and gates {w>>2, (w>>2)+2} -> shared A frag.
// Weights (4 gates x 16 cols x 384 K = 48KB bf16) TDM-loaded into LDS once.
__global__ __launch_bounds__(256)
void lstm_persistent_kernel(const __bf16* __restrict__ xb,    // (S,B,D)
                            const __bf16* __restrict__ Wp,    // (NF,KTOT) W^T
                            const float* __restrict__ bf_, const float* __restrict__ bi_,
                            const float* __restrict__ bg_, const float* __restrict__ bo_,
                            const int*   __restrict__ mask,   // (S,B)
                            __bf16* __restrict__ hbuf0, __bf16* __restrict__ hbuf1,
                            float*  __restrict__ out,         // (S,B,H) + [hx|cx]
                            unsigned* __restrict__ bar) {
  __shared__ __bf16 wlds[64 * KTOT];   // 48 KB: row n_local = g*16 + col
  __shared__ float  gsm[16][256];      // 16 KB: tile t = g*4 + bt

  const int tid  = threadIdx.x;
  const int lane = tid & 31;
  const int w    = tid >> 5;
  const int ht   = blockIdx.x & 15;
  const int bg   = blockIdx.x >> 4;
  const int h0   = ht * 16;
  const int gA   = w >> 2;             // gates gA and gA+2
  const int bt   = w & 3;
  const int rowb = bg * 64 + bt * 16;

  // ---- TDM: pull this block's weight slab into LDS (wave 0 only) ----
  if (tid < 32) {
    unsigned lds_base =
        (unsigned)(uintptr_t)(__attribute__((address_space(3))) __bf16*)wlds;
    unsigned long long ga = (unsigned long long)(uintptr_t)(Wp + (size_t)h0 * KTOT);
    u32x4 g0;
    g0[0] = 1u;                                            // count=1, user mode
    g0[1] = lds_base;                                      // lds_addr
    g0[2] = (unsigned)(ga & 0xFFFFFFFFull);                // global_addr[31:0]
    g0[3] = (unsigned)((ga >> 32) & 0x1FFFFFFull) | (2u << 30); // addr[56:32], type=2
    i32x8 g1;
    g1[0] = 0x00010000;                                    // data_size=1 (2 bytes)
    g1[1] = (int)((6144u & 0xFFFFu) << 16);                // tensor_dim0 = 6144 (lo16)
    g1[2] = (int)(4u << 16);                               // tdim0 hi=0 | tensor_dim1=4
    g1[3] = (int)(6144u << 16);                            // tdim1 hi=0 | tile_dim0=6144
    g1[4] = 4;                                             // tile_dim1=4, tile_dim2=0
    g1[5] = (int)98304u;                                   // tensor_dim0_stride = 256*384
    g1[6] = 0;
    g1[7] = 0;
    i32x4 z4 = {0, 0, 0, 0};
    i32x8 z8 = {0, 0, 0, 0, 0, 0, 0, 0};
    __builtin_amdgcn_tensor_load_to_lds(g0, g1, z4, z4, z8, 0);
    __builtin_amdgcn_s_wait_tensorcnt(0);
  }
  __syncthreads();

  float creg[4] = {0.f, 0.f, 0.f, 0.f};   // cell state lives in registers

  for (int s = 0; s < S_; ++s) {
    const __bf16* hp = (s & 1) ? hbuf1 : hbuf0;
    __bf16*       hn = (s & 1) ? hbuf0 : hbuf1;

    v8f acc0 = {}, acc1 = {};
    // x part: K = 0..127
    const __bf16* xbase = xb + (size_t)s * B_ * D_ + (size_t)rowb * D_;
#pragma unroll
    for (int kb = 0; kb < 4; ++kb) {
      v16bf a  = load_a_bf16(xbase + kb * 32, D_, lane);
      v16bf b0 = load_b_rowmajT(wlds, KTOT, gA * 16,       kb, lane);
      v16bf b1 = load_b_rowmajT(wlds, KTOT, (gA + 2) * 16, kb, lane);
      acc0 = wmma_bf16(a, b0, acc0);
      acc1 = wmma_bf16(a, b1, acc1);
    }
    // h part: K = 128..383
    const __bf16* hbase = hp + (size_t)rowb * H_;
#pragma unroll
    for (int kb = 0; kb < 8; ++kb) {
      v16bf a  = load_a_bf16(hbase + kb * 32, H_, lane);
      v16bf b0 = load_b_rowmajT(wlds, KTOT, gA * 16,       4 + kb, lane);
      v16bf b1 = load_b_rowmajT(wlds, KTOT, (gA + 2) * 16, 4 + kb, lane);
      acc0 = wmma_bf16(a, b0, acc0);
      acc1 = wmma_bf16(a, b1, acc1);
    }
    // C layout: VGPR r -> M = r + 8*(lane>=16), N = lane&15
    const int t0 = gA * 4 + bt, t1 = t0 + 8;
#pragma unroll
    for (int r = 0; r < 8; ++r) {
      int m = r + ((lane & 16) ? 8 : 0);
      gsm[t0][m * 16 + (lane & 15)] = acc0[r];
      gsm[t1][m * 16 + (lane & 15)] = acc1[r];
    }
    __syncthreads();

    // elementwise LSTM cell update: 64x16 elements / 256 threads = 4 each
#pragma unroll
    for (int j = 0; j < 4; ++j) {
      int e   = j * 256 + tid;
      int bl  = e >> 4, hh = e & 15;
      int btj = bl >> 4, mm = bl & 15;
      int b   = bg * 64 + bl;
      int h   = h0 + hh;
      float mk = (float)mask[(size_t)s * B_ + b];
      float fg = sigmoidf_(gsm[0 * 4 + btj][mm * 16 + hh] + bf_[h]);
      float ig = sigmoidf_(gsm[1 * 4 + btj][mm * 16 + hh] + bi_[h]);
      float gg = tanhf   (gsm[2 * 4 + btj][mm * 16 + hh] + bg_[h]);
      float og = sigmoidf_(gsm[3 * 4 + btj][mm * 16 + hh] + bo_[h]);
      fg = fg * mk + (1.f - mk);
      ig *= mk;
      gg *= mk;
      og = og * mk + (1.f - mk);
      float c  = fg * creg[j] + ig * gg;
      float hv = og * tanhf(c);
      creg[j] = c;
      size_t idx = (size_t)b * H_ + h;
      hn[idx] = (__bf16)hv;
      out[(size_t)s * B_ * H_ + idx] = hv;
      if (s == S_ - 1) {
        out[(size_t)S_ * B_ * H_ + idx] = hv;                       // hx
        out[(size_t)S_ * B_ * H_ + (size_t)B_ * H_ + idx] = c;      // cx
      }
    }

    // ---- grid-wide barrier between timesteps ----
    __syncthreads();
    if (tid == 0) {
      __threadfence();
      unsigned gen = __hip_atomic_load(&bar[1], __ATOMIC_RELAXED, __HIP_MEMORY_SCOPE_AGENT);
      unsigned arr = __hip_atomic_fetch_add(&bar[0], 1u, __ATOMIC_ACQ_REL,
                                            __HIP_MEMORY_SCOPE_AGENT) + 1u;
      if (arr == NBLK) {
        __hip_atomic_store(&bar[0], 0u, __ATOMIC_RELAXED, __HIP_MEMORY_SCOPE_AGENT);
        __hip_atomic_store(&bar[1], gen + 1u, __ATOMIC_RELEASE, __HIP_MEMORY_SCOPE_AGENT);
      } else {
        while (__hip_atomic_load(&bar[1], __ATOMIC_ACQUIRE, __HIP_MEMORY_SCOPE_AGENT) == gen)
          __builtin_amdgcn_s_sleep(1);
      }
    }
    __syncthreads();
  }
}

// ---- attention: online softmax over S, chunked ------------------------------

__global__ __launch_bounds__(32)
void attn_pass1_kernel(const float* __restrict__ out,   // (S,B,H) h history
                       const __bf16* __restrict__ Wpa,  // (A,H) Wa^T
                       const float* __restrict__ ba,
                       float* __restrict__ pm, float* __restrict__ pl,
                       float* __restrict__ pa) {
  const int lane = threadIdx.x & 31;
  const int b0 = blockIdx.x * 16;
  const int a0 = blockIdx.y * 16;
  const int ch = blockIdx.z;

  v16bf wb[8];
#pragma unroll
  for (int kb = 0; kb < 8; ++kb) wb[kb] = load_b_rowmajT(Wpa, H_, a0, kb, lane);

  const float bias = ba[a0 + (lane & 15)];
  const int   mrow = (lane & 16) ? 8 : 0;
  const int   acol = a0 + (lane & 15);

  float m[8], l[8], acc[8];
#pragma unroll
  for (int r = 0; r < 8; ++r) { m[r] = -3.0e38f; l[r] = 0.f; acc[r] = 0.f; }

  for (int s = ch * SCH; s < (ch + 1) * SCH; ++s) {
    const float* base = out + (size_t)s * B_ * H_ + (size_t)b0 * H_;
    v8f sc = {};
#pragma unroll
    for (int kb = 0; kb < 8; ++kb) {
      v16bf a = load_a_f32(base + kb * 32, H_, lane);
      sc = wmma_bf16(a, wb[kb], sc);
    }
#pragma unroll
    for (int r = 0; r < 8; ++r) {
      int b = b0 + r + mrow;
      float score = sc[r] + bias;
      float v  = out[(size_t)s * B_ * H_ + (size_t)b * H_ + acol];
      float mn = fmaxf(m[r], score);
      float scl = __expf(m[r] - mn);
      float e   = __expf(score - mn);
      l[r]   = l[r] * scl + e;
      acc[r] = acc[r] * scl + e * v;
      m[r]   = mn;
    }
  }
#pragma unroll
  for (int r = 0; r < 8; ++r) {
    int b = b0 + r + mrow;
    size_t idx = ((size_t)ch * B_ + b) * A_ + acol;
    pm[idx] = m[r]; pl[idx] = l[r]; pa[idx] = acc[r];
  }
}

__global__ void attn_combine_kernel(const float* __restrict__ pm,
                                    const float* __restrict__ pl,
                                    const float* __restrict__ pa,
                                    float* __restrict__ ctx) {
  int i = blockIdx.x * blockDim.x + threadIdx.x;
  if (i >= B_ * A_) return;
  float M = -3.0e38f;
#pragma unroll
  for (int c = 0; c < NCH; ++c) M = fmaxf(M, pm[(size_t)c * B_ * A_ + i]);
  float L = 0.f, ACC = 0.f;
#pragma unroll
  for (int c = 0; c < NCH; ++c) {
    float w = __expf(pm[(size_t)c * B_ * A_ + i] - M);
    L   += pl[(size_t)c * B_ * A_ + i] * w;
    ACC += pa[(size_t)c * B_ * A_ + i] * w;
  }
  ctx[i] = ACC / L;
}

__global__ void broadcast_kernel(const float* __restrict__ ctx, float* __restrict__ out) {
  int i = blockIdx.x * blockDim.x + threadIdx.x; // over B*H
  int s = blockIdx.y;
  out[(size_t)s * B_ * H_ + i] = ctx[i];
}

// ---- launch -----------------------------------------------------------------

extern "C" void kernel_launch(void* const* d_in, const int* in_sizes, int n_in,
                              void* d_out, int out_size, void* d_ws, size_t ws_size,
                              hipStream_t stream) {
  (void)in_sizes; (void)n_in; (void)out_size; (void)ws_size;
  const float* x    = (const float*)d_in[0];
  const int*   mask = (const int*)d_in[1];
  const float* Wf = (const float*)d_in[2];  const float* bf_ = (const float*)d_in[3];
  const float* Wi = (const float*)d_in[4];  const float* bi_ = (const float*)d_in[5];
  const float* Wg = (const float*)d_in[6];  const float* bg_ = (const float*)d_in[7];
  const float* Wo = (const float*)d_in[8];  const float* bo_ = (const float*)d_in[9];
  const float* Wa = (const float*)d_in[10]; const float* ba  = (const float*)d_in[11];
  float* out = (float*)d_out;

  // workspace layout (~73 MB)
  char* w = (char*)d_ws;
  size_t off = 0;
  auto alloc = [&](size_t bytes) -> void* {
    void* p = w + off;
    off = (off + bytes + 255) & ~(size_t)255;
    return p;
  };
  __bf16*   xb  = (__bf16*)alloc((size_t)S_ * B_ * D_ * 2);
  __bf16*   Wp  = (__bf16*)alloc((size_t)NF * KTOT * 2);
  __bf16*   Wpa = (__bf16*)alloc((size_t)A_ * H_ * 2);
  __bf16*   hb0 = (__bf16*)alloc((size_t)B_ * H_ * 2);
  __bf16*   hb1 = (__bf16*)alloc((size_t)B_ * H_ * 2);
  unsigned* bar = (unsigned*)alloc(256);
  float*    pm  = (float*)alloc((size_t)NCH * B_ * A_ * 4);
  float*    pl  = (float*)alloc((size_t)NCH * B_ * A_ * 4);
  float*    pa  = (float*)alloc((size_t)NCH * B_ * A_ * 4);
  float*    ctx = (float*)alloc((size_t)B_ * H_ * 4);

  { long n = (long)S_ * B_ * D_;
    cvt_x_kernel<<<(unsigned)((n + 255) / 256), 256, 0, stream>>>(x, xb, n); }
  pack_gates_kernel<<<(NF * KTOT + 255) / 256, 256, 0, stream>>>(Wf, Wi, Wg, Wo, Wp);
  pack_wa_kernel<<<(A_ * H_ + 255) / 256, 256, 0, stream>>>(Wa, Wpa);
  init_state_kernel<<<(B_ * H_ + 255) / 256, 256, 0, stream>>>(hb0, bar);

  lstm_persistent_kernel<<<NBLK, 256, 0, stream>>>(xb, Wp, bf_, bi_, bg_, bo_, mask,
                                                   hb0, hb1, out, bar);

  dim3 agrid(B_ / 16, A_ / 16, NCH);
  attn_pass1_kernel<<<agrid, 32, 0, stream>>>(out, Wpa, ba, pm, pl, pa);
  attn_combine_kernel<<<(B_ * A_ + 255) / 256, 256, 0, stream>>>(pm, pl, pa, ctx);
  dim3 bgrid(B_ * H_ / 256, S_);
  broadcast_kernel<<<bgrid, 256, 0, stream>>>(ctx, out);
}
